// inception_block_dcn_v2_58815282152117
// MI455X (gfx1250) — compile-verified, matrix operand order checked by reading, not run on hardware
//
#include <hip/hip_runtime.h>
#include <hip/hip_bf16.h>

// ---------------------------------------------------------------------------
// MI455X (gfx1250) inception+DCNv2 block.
//  - All GEMM work on v_wmma_f32_16x16x32_f16 (wave32), f32 accumulate.
//  - Activations NHWC f16 via fused BN+ReLU epilogues.
//  - Weights pre-packed to f16 [tap][CoutPad][Cin]; WMMA A-fragments loaded
//    directly from global (L2-hot), no LDS staging for A.
//  - 1x1 convs (pad-free) stage their whole B panel with the Tensor Data
//    Mover (tensor_load_to_lds + s_wait_tensorcnt).
//  - DCN gather: clamped-coordinate, validity-folded-into-weights, 16B-wide
//    unconditional corner loads (no exec-mask branching).
// ---------------------------------------------------------------------------

typedef _Float16 h16;
typedef _Float16 v8h  __attribute__((ext_vector_type(8)));
typedef _Float16 v16h __attribute__((ext_vector_type(16)));
typedef float    v8f  __attribute__((ext_vector_type(8)));
typedef unsigned int u32;
typedef u32 v4u  __attribute__((ext_vector_type(4)));
typedef int v8i_t __attribute__((ext_vector_type(8)));
typedef int v4i_t __attribute__((ext_vector_type(4)));

#define HH    128
#define WW    128
#define HWSZ  16384          // H*W
#define NP    65536          // N*H*W
#define EPSF  1e-5f

// ---- WMMA fragment loader (16-bit A/B operand, 16x32 tile) ----------------
// lane L (half=L>>4, r=L&15): row M=r, two contiguous 8-halfword runs at
// K = half*8 and K = 16+half*8. Works for A ([M][K]) and B stored [N][K].
static __device__ inline v16h load_frag(const h16* base, int rowStride,
                                        int lane) {
  const int r = lane & 15;
  const int half = lane >> 4;
  v8h lo = *(const v8h*)(base + (long)r * rowStride + half * 8);
  v8h hi = *(const v8h*)(base + (long)r * rowStride + 16 + half * 8);
  return __builtin_shufflevector(lo, hi, 0, 1, 2, 3, 4, 5, 6, 7,
                                 8, 9, 10, 11, 12, 13, 14, 15);
}

static __device__ inline v8f wmma_f16(v16h a, v16h b, v8f c) {
  return __builtin_amdgcn_wmma_f32_16x16x32_f16(false, a, false, b,
                                                (short)0, c, false, false);
}

// ---- shared epilogue ------------------------------------------------------
// D layout: VGPR g -> M = g + 8*(lane>=16), N = lane&15.
// MODE 0: next-layer BN+ReLU -> NHWC f16; MODE 1: +bias, sigmoid(ch>=18),
// NHWC f32 stride 32; MODE 2: plain f32 NCHW.
template <int MODE>
static __device__ inline void conv_epilogue(
    const v8f& acc, int coB, int pB, int lane, int Cout, int pbase,
    const float* p0, const float* p1, const float* p2, const float* p3,
    void* outp, int outStride, int outOff) {
  const int r = lane & 15, half = lane >> 4;
  const int p = pbase + pB * 16 + r;
#pragma unroll
  for (int g = 0; g < 8; ++g) {
    const int co = coB * 16 + g + half * 8;
    if (co >= Cout) continue;
    float val = acc[g];
    if (MODE == 0) {
      float inv = p0[co] * rsqrtf(p3[co] + EPSF);
      val = fmaxf(val * inv + (p1[co] - p2[co] * inv), 0.f);
      ((h16*)outp)[(long)p * outStride + outOff + co] = (h16)val;
    } else if (MODE == 1) {
      val += p0[co];
      if (co >= 18) val = 1.f / (1.f + expf(-val));
      ((float*)outp)[(long)p * 32 + co] = val;
    } else {
      const int nn = p >> 14, hw = p & (HWSZ - 1);
      ((float*)outp)[(long)(nn * Cout + co) * HWSZ + hw] = val;
    }
  }
}

// ---- kernel: BN+ReLU on NCHW f32 input -> NHWC f16 ------------------------
__global__ __launch_bounds__(256) void act_kernel(
    const float* __restrict__ x, const float* __restrict__ g,
    const float* __restrict__ b, const float* __restrict__ m,
    const float* __restrict__ v, h16* __restrict__ out) {
  int i = blockIdx.x * blockDim.x + threadIdx.x;   // NCHW order
  if (i >= 4 * 128 * HWSZ) return;
  int hw = i & (HWSZ - 1);
  int c  = (i >> 14) & 127;
  int n  = i >> 21;
  float inv = g[c] * rsqrtf(v[c] + EPSF);
  float val = fmaxf(x[i] * inv + (b[c] - m[c] * inv), 0.f);
  out[((n * HWSZ + hw) << 7) + c] = (h16)val;
}

// ---- kernel: weight pre-pack f32 OIHW -> f16 [tap][CoutPad][Cin] ----------
__global__ __launch_bounds__(256) void pack_w(
    const float* __restrict__ w, h16* __restrict__ out,
    int Cout, int Cin, int KH, int KW, int CoutPad) {
  int i = blockIdx.x * 256 + threadIdx.x;
  int total = KH * KW * CoutPad * Cin;
  if (i >= total) return;
  int ci = i % Cin;
  int t = i / Cin;
  int co = t % CoutPad;
  int tap = t / CoutPad;
  int ty = tap / KW, tx = tap % KW;
  float v = (co < Cout) ? w[((long)(co * Cin + ci) * KH + ty) * KW + tx] : 0.f;
  out[i] = (h16)v;
}

// ---- kernel: generic tap conv (1xk, kx1, 3x3) -----------------------------
// B staged in LDS (zero-padded taps); A fragments straight from packed global.
template <int MODE>
__global__ __launch_bounds__(256) void conv_wmma(
    const h16* __restrict__ act, const h16* __restrict__ w16,
    int Cin, int Cout, int CoutPad, int KH, int KW, int padH, int padW,
    const float* __restrict__ p0, const float* __restrict__ p1,
    const float* __restrict__ p2, const float* __restrict__ p3,
    void* __restrict__ outp, int outStride, int outOff) {
  __shared__ __align__(16) h16 sB[64 * 32];

  const int tid = threadIdx.x, lane = tid & 31, wid = tid >> 5;
  const int nCo = (Cout + 15) >> 4;
  const int T = nCo * 4;
  const int pbase = blockIdx.x * 64;
  const int n = pbase >> 14, h = (pbase >> 7) & 127, w0 = pbase & 127;

  v8f acc0 = {0.f, 0.f, 0.f, 0.f, 0.f, 0.f, 0.f, 0.f};
  v8f acc1 = acc0;
  const int s0 = wid, s1 = wid + 8;
  const int pl = tid >> 2;           // B-stage position row
  const int cf = (tid & 3) * 8;      // B-stage channel chunk

  for (int ty = 0; ty < KH; ++ty) {
    const int hs = h + ty - padH;
    if ((unsigned)hs >= (unsigned)HH) continue;   // uniform: tap row all-pad
    for (int tx = 0; tx < KW; ++tx) {
      const int ws = w0 + tx - padW;
      const int tap = ty * KW + tx;
      for (int cb = 0; cb < Cin; cb += 32) {
        {
          const int wsl = ws + pl;
          v8h bv;
#pragma unroll
          for (int q = 0; q < 8; ++q) bv[q] = (h16)0.f;
          if ((unsigned)wsl < (unsigned)WW)
            bv = *(const v8h*)(act +
                               (long)((n * HH + hs) * WW + wsl) * Cin + cb + cf);
          *(v8h*)(sB + pl * 32 + cf) = bv;
        }
        __syncthreads();
        if (s0 < T) {
          const int coB = s0 % nCo, pB = s0 / nCo;
          v16h a = load_frag(w16 + ((long)tap * CoutPad + coB * 16) * Cin + cb,
                             Cin, lane);
          v16h b = load_frag(sB + pB * 512, 32, lane);
          acc0 = wmma_f16(a, b, acc0);
        }
        if (s1 < T) {
          const int coB = s1 % nCo, pB = s1 / nCo;
          v16h a = load_frag(w16 + ((long)tap * CoutPad + coB * 16) * Cin + cb,
                             Cin, lane);
          v16h b = load_frag(sB + pB * 512, 32, lane);
          acc1 = wmma_f16(a, b, acc1);
        }
        __syncthreads();
      }
    }
  }
  if (s0 < T)
    conv_epilogue<MODE>(acc0, s0 % nCo, s0 / nCo, lane, Cout, pbase,
                        p0, p1, p2, p3, outp, outStride, outOff);
  if (s1 < T)
    conv_epilogue<MODE>(acc1, s1 % nCo, s1 / nCo, lane, Cout, pbase,
                        p0, p1, p2, p3, outp, outStride, outOff);
}

// ---- kernel: pad-free 1x1 conv, B panel staged by the Tensor Data Mover ---
// One TDM 2D-tile load (64 positions x Cin, row stride Cin) fills LDS once;
// K-loop then runs barrier-free. CoutPad == 64 for all users (Cout == 64).
template <int MODE>
__global__ __launch_bounds__(256) void conv1x1_tdm(
    const h16* __restrict__ act, const h16* __restrict__ w16, int Cin,
    const float* __restrict__ p0, const float* __restrict__ p1,
    const float* __restrict__ p2, const float* __restrict__ p3,
    void* __restrict__ outp, int outStride, int outOff) {
  __shared__ __align__(16) h16 sB[64 * 256];   // up to Cin=256 (32 KB)

  const int tid = threadIdx.x, lane = tid & 31, wid = tid >> 5;
  const int pbase = blockIdx.x * 64;

#if defined(__gfx1250__) && __has_builtin(__builtin_amdgcn_tensor_load_to_lds)
  if (wid == 0) {
    const unsigned long long ga =
        (unsigned long long)(const void*)(act + (long)pbase * Cin);
    const u32 lds = (u32)(unsigned long long)(void*)sB;
    v4u g0;
    g0[0] = 1u;                                  // count=1, user mode
    g0[1] = lds;                                 // lds_addr
    g0[2] = (u32)ga;                             // global_addr[31:0]
    g0[3] = (u32)(ga >> 32) | (2u << 30);        // global_addr[56:32] | type=2
    v8i_t g1;
    g1[0] = 1 << 16;                             // data_size=1 (2B), mask=0
    g1[1] = (int)(((u32)Cin & 0xFFFFu) << 16);   // tensor_dim0[15:0]
    g1[2] = (int)(64u << 16);                    // dim0[31:16]=0 | tensor_dim1=64
    g1[3] = (int)(((u32)Cin & 0xFFFFu) << 16);   // dim1[31:16]=0 | tile_dim0=Cin
    g1[4] = 64;                                  // tile_dim1=64, tile_dim2=0
    g1[5] = Cin;                                 // tensor_dim0_stride[31:0]
    g1[6] = 0;
    g1[7] = 0;
    v4i_t z4 = {0, 0, 0, 0};
    v8i_t z8 = {0, 0, 0, 0, 0, 0, 0, 0};
    __builtin_amdgcn_tensor_load_to_lds(g0, g1, z4, z4, z8, 0);
    __builtin_amdgcn_s_wait_tensorcnt(0);
  }
#else
  for (int idx = tid; idx < (64 * Cin) / 8; idx += 256)
    *(v8h*)(sB + idx * 8) = *(const v8h*)(act + (long)pbase * Cin + idx * 8);
#endif
  __syncthreads();

  v8f acc0 = {0.f, 0.f, 0.f, 0.f, 0.f, 0.f, 0.f, 0.f};
  v8f acc1 = acc0;
  const int coB0 = wid & 3, pB0 = wid >> 2;       // tiles wid and wid+8
  const int coB1 = coB0, pB1 = pB0 + 2;
  for (int cb = 0; cb < Cin; cb += 32) {
    v16h a = load_frag(w16 + (long)(coB0 * 16) * Cin + cb, Cin, lane);
    v16h b0 = load_frag(sB + (long)(pB0 * 16) * Cin + cb, Cin, lane);
    v16h b1 = load_frag(sB + (long)(pB1 * 16) * Cin + cb, Cin, lane);
    acc0 = wmma_f16(a, b0, acc0);
    acc1 = wmma_f16(a, b1, acc1);
  }
  conv_epilogue<MODE>(acc0, coB0, pB0, lane, 64, pbase, p0, p1, p2, p3,
                      outp, outStride, outOff);
  conv_epilogue<MODE>(acc1, coB1, pB1, lane, 64, pbase, p0, p1, p2, p3,
                      outp, outStride, outOff);
}

// ---- kernel: DCNv2 implicit GEMM ------------------------------------------
// 16-position tile; per tap: clamped-coordinate bilinear gather (validity in
// the corner weights -> unconditional 16B loads), one b128 LDS store per
// thread, then 2 WMMAs per wave against pre-packed f16 weights from global.
__global__ __launch_bounds__(128) void dcn_wmma_kernel(
    const h16* __restrict__ yact, const float* __restrict__ om,
    const h16* __restrict__ dw16 /*[9][64][64]*/,
    const float* __restrict__ cg, const float* __restrict__ cbb,
    const float* __restrict__ cm, const float* __restrict__ cv,
    int chOff, h16* __restrict__ catbuf) {
  __shared__ __align__(16) h16 sB[16 * 64];

  const int tid = threadIdx.x, lane = tid & 31, wid = tid >> 5;
  const int pbase = blockIdx.x * 16;
  const int pidx = tid & 15;
  const int p = pbase + pidx;
  const int n = p >> 14, h = (p >> 7) & 127, w = p & 127;
  const int c0 = (tid >> 4) * 8;          // this thread's 8 channels

  v8f acc = {0.f, 0.f, 0.f, 0.f, 0.f, 0.f, 0.f, 0.f};

  for (int k = 0; k < 9; ++k) {
    const int ky = k / 3 - 1, kx = k % 3 - 1;
    const float* ob = om + (long)p * 32;
    const float oy = ob[2 * k], ox = ob[2 * k + 1], msk = ob[18 + k];
    const float py = (float)(h + ky) + oy;
    const float px = (float)(w + kx) + ox;
    const float fy = floorf(py), fx = floorf(px);
    const int y0 = (int)fy, x0 = (int)fx;
    const float wy = py - fy, wx = px - fx;
    // validity folded into weights; coordinates clamped -> in-bounds loads
    const float vy0 = (y0 >= 0 && y0 < HH) ? 1.f : 0.f;
    const float vy1 = (y0 + 1 >= 0 && y0 + 1 < HH) ? 1.f : 0.f;
    const float vx0 = (x0 >= 0 && x0 < WW) ? 1.f : 0.f;
    const float vx1 = (x0 + 1 >= 0 && x0 + 1 < WW) ? 1.f : 0.f;
    const float w00 = (1.f - wy) * (1.f - wx) * msk * vy0 * vx0;
    const float w01 = (1.f - wy) * wx * msk * vy0 * vx1;
    const float w10 = wy * (1.f - wx) * msk * vy1 * vx0;
    const float w11 = wy * wx * msk * vy1 * vx1;
    const int y0c = min(max(y0, 0), HH - 1), y1c = min(max(y0 + 1, 0), HH - 1);
    const int x0c = min(max(x0, 0), WW - 1), x1c = min(max(x0 + 1, 0), WW - 1);
    const long r0 = (long)(n * HH + y0c) * WW;
    const long r1 = (long)(n * HH + y1c) * WW;
    v8h q00 = *(const v8h*)(yact + (r0 + x0c) * 64 + c0);
    v8h q01 = *(const v8h*)(yact + (r0 + x1c) * 64 + c0);
    v8h q10 = *(const v8h*)(yact + (r1 + x0c) * 64 + c0);
    v8h q11 = *(const v8h*)(yact + (r1 + x1c) * 64 + c0);
    v8h res;
#pragma unroll
    for (int j = 0; j < 8; ++j) {
      float s = w00 * (float)q00[j] + w01 * (float)q01[j] +
                w10 * (float)q10[j] + w11 * (float)q11[j];
      res[j] = (h16)s;
    }
    __syncthreads();                       // prev tap's reads done
    *(v8h*)(sB + pidx * 64 + c0) = res;
    __syncthreads();                       // tile visible
#pragma unroll
    for (int cb = 0; cb < 64; cb += 32) {
      v16h a = load_frag(dw16 + (long)(k * 64 + wid * 16) * 64 + cb, 64, lane);
      v16h b = load_frag(sB + cb, 64, lane);
      acc = wmma_f16(a, b, acc);
    }
  }

  const int r = lane & 15, half = lane >> 4;
  const int pp = pbase + r;
#pragma unroll
  for (int g = 0; g < 8; ++g) {
    const int co = wid * 16 + g + half * 8;
    const int ch = chOff + co;
    float inv = cg[ch] * rsqrtf(cv[ch] + EPSF);
    float val = fmaxf(acc[g] * inv + (cbb[ch] - cm[ch] * inv), 0.f);
    catbuf[(long)pp * 256 + ch] = (h16)val;
  }
}

// ---------------------------------------------------------------------------
extern "C" void kernel_launch(void* const* d_in, const int* in_sizes, int n_in,
                              void* d_out, int out_size, void* d_ws,
                              size_t ws_size, hipStream_t stream) {
  (void)in_sizes; (void)n_in; (void)out_size; (void)ws_size;
  const float* X = (const float*)d_in[0];

  char* ws = (char*)d_ws;
  auto carve = [&](size_t bytes) {
    char* r = ws;
    ws += (bytes + 255) & ~(size_t)255;
    return r;
  };
  h16*   a0    = (h16*)carve((size_t)NP * 128 * 2);
  h16*   t1    = (h16*)carve((size_t)NP * 64 * 2);
  h16*   t2    = (h16*)carve((size_t)NP * 64 * 2);
  h16*   yact  = (h16*)carve((size_t)NP * 64 * 2);
  float* om    = (float*)carve((size_t)NP * 32 * 4);
  h16*   catb  = (h16*)carve((size_t)NP * 256 * 2);
  h16*   pw1   = (h16*)carve(64 * 128 * 2);
  h16*   pw2   = (h16*)carve(7 * 64 * 64 * 2);
  h16*   pw3   = (h16*)carve(7 * 64 * 64 * 2);
  h16*   pwoff = (h16*)carve(9 * 32 * 64 * 2);
  h16*   pwdcn = (h16*)carve(9 * 64 * 64 * 2);
  h16*   pwcat = (h16*)carve(64 * 256 * 2);

  struct BIdx { int c1, c1w, c2, c2w, c3, c3w, bn, offw, offb, dcnw, k; };
  const BIdx B[4] = {
      {1, 5, 6, 10, -1, -1, 11, 15, 16, 17, 1},
      {18, 22, 23, 27, 28, 32, 33, 37, 38, 39, 3},
      {40, 44, 45, 49, 50, 54, 55, 59, 60, 61, 5},
      {62, 66, 67, 71, 72, 76, 77, 81, 82, 83, 7},
  };
  const int catg = 84, catw = 88;
#define F(i) ((const float*)d_in[(i)])
  auto packGrid = [](int total) { return dim3((total + 255) / 256); };

  dim3 gConv(NP / 64), bConv(256);
  dim3 gAct((4 * 128 * HWSZ) / 256), bAct(256);
  dim3 gDcn(NP / 16), bDcn(128);

  for (int bi = 0; bi < 4; ++bi) {
    const BIdx& I = B[bi];
    act_kernel<<<gAct, bAct, 0, stream>>>(X, F(I.c1), F(I.c1 + 1),
                                          F(I.c1 + 2), F(I.c1 + 3), a0);
    // c1: 1x1 128->64 (TDM), epilogue BN(c2)+ReLU
    pack_w<<<packGrid(64 * 128), 256, 0, stream>>>(F(I.c1w), pw1, 64, 128, 1, 1, 64);
    conv1x1_tdm<0><<<gConv, bConv, 0, stream>>>(
        a0, pw1, 128, F(I.c2), F(I.c2 + 1), F(I.c2 + 2), F(I.c2 + 3), t1, 64, 0);
    if (I.c3 < 0) {
      pack_w<<<packGrid(64 * 64), 256, 0, stream>>>(F(I.c2w), pw2, 64, 64, 1, 1, 64);
      conv1x1_tdm<0><<<gConv, bConv, 0, stream>>>(
          t1, pw2, 64, F(I.bn), F(I.bn + 1), F(I.bn + 2), F(I.bn + 3), yact, 64, 0);
    } else {
      const int k = I.k, pw = (k - 1) / 2;
      pack_w<<<packGrid(k * 64 * 64), 256, 0, stream>>>(F(I.c2w), pw2, 64, 64, 1, k, 64);
      conv_wmma<0><<<gConv, bConv, 0, stream>>>(
          t1, pw2, 64, 64, 64, 1, k, 0, pw,
          F(I.c3), F(I.c3 + 1), F(I.c3 + 2), F(I.c3 + 3), t2, 64, 0);
      pack_w<<<packGrid(k * 64 * 64), 256, 0, stream>>>(F(I.c3w), pw3, 64, 64, k, 1, 64);
      conv_wmma<0><<<gConv, bConv, 0, stream>>>(
          t2, pw3, 64, 64, 64, k, 1, pw, 0,
          F(I.bn), F(I.bn + 1), F(I.bn + 2), F(I.bn + 3), yact, 64, 0);
    }
    // offset conv 3x3 64->27 (pad Cout to 32), epilogue bias+sigmoid
    pack_w<<<packGrid(9 * 32 * 64), 256, 0, stream>>>(F(I.offw), pwoff, 27, 64, 3, 3, 32);
    conv_wmma<1><<<gConv, bConv, 0, stream>>>(
        yact, pwoff, 64, 27, 32, 3, 3, 1, 1,
        F(I.offb), nullptr, nullptr, nullptr, om, 32, 0);
    // DCNv2 + cat-BN slice + ReLU -> interleaved concat buffer
    pack_w<<<packGrid(9 * 64 * 64), 256, 0, stream>>>(F(I.dcnw), pwdcn, 64, 64, 3, 3, 64);
    dcn_wmma_kernel<<<gDcn, bDcn, 0, stream>>>(
        yact, om, pwdcn, F(catg), F(catg + 1), F(catg + 2), F(catg + 3),
        bi * 64, catb);
  }
  // final 1x1 256->64 (TDM), plain f32 NCHW output
  pack_w<<<packGrid(64 * 256), 256, 0, stream>>>(F(catw), pwcat, 64, 256, 1, 1, 64);
  conv1x1_tdm<2><<<gConv, bConv, 0, stream>>>(
      catb, pwcat, 256, nullptr, nullptr, nullptr, nullptr, d_out, 0, 0);
#undef F
}
